// Encoder_36764920054419
// MI455X (gfx1250) — compile-verified, matrix operand order checked by reading, not run on hardware
//
#include <hip/hip_runtime.h>
#include <hip/hip_bf16.h>

// MI455X / gfx1250 implementation.
// bf16 activations (L2-resident ping-pong buffers), f32 WMMA accumulation via
// v_wmma_f32_16x16x32_bf16, implicit-GEMM conv / conv-transpose with fully
// compile-time-specialized layer shapes, block-cooperative double-buffered
// weight staging into LDS via the Tensor Data Mover (tensor_load_to_lds +
// s_wait_tensorcnt, with a cooperative-copy fallback), GN folded to per-(b,c)
// scale+shift with coalesced vectorized reductions, LDS-atomic instance pool.

typedef __attribute__((ext_vector_type(16))) __bf16       v16bf;
typedef __attribute__((ext_vector_type(8)))  __bf16       v8bf;
typedef __attribute__((ext_vector_type(8)))  float        v8f;
typedef __attribute__((ext_vector_type(4)))  unsigned int u32x4;
typedef __attribute__((ext_vector_type(8)))  int          i32x8;
typedef __attribute__((ext_vector_type(4)))  int          i32x4;

#define NUM_INST 32

#if defined(__has_builtin)
#if __has_builtin(__builtin_amdgcn_tensor_load_to_lds) && \
    __has_builtin(__builtin_amdgcn_s_wait_tensorcnt)
#define HAVE_TDM 1
#endif
#endif

// K index held by fragment element e for lane-half h (16-bit A/B layout, ISA 7.12.2)
__device__ __forceinline__ int kmap(int e, int half) {
  return e + 8 * half + ((e >= 8) ? 8 : 0);
}

// ---------------------------------------------------------------------------
// Pack f32 HWIO weights into the exact per-lane B-fragment layout:
// out[((chunk*ntile + t)*32 + lane)*16 + e] = bf16(W[k, n]) with
// k = chunk*32 + kmap(e, lane>>4), n = t*16 + (lane&15), zero-padded.
// ---------------------------------------------------------------------------
__global__ void pack_weights(const float* __restrict__ w, __bf16* __restrict__ out,
                             int K, int CO, int nchunk, int ntile) {
  int idx = blockIdx.x * blockDim.x + threadIdx.x;
  int total = nchunk * ntile * 32 * 16;
  if (idx >= total) return;
  int e    = idx & 15;
  int lane = (idx >> 4) & 31;
  int r    = idx >> 9;
  int t    = r % ntile;
  int c    = r / ntile;
  int n    = t * 16 + (lane & 15);
  int k    = c * 32 + kmap(e, lane >> 4);
  float v = 0.f;
  if (k < K && n < CO) v = w[(size_t)k * CO + n];
  out[idx] = (__bf16)v;
}

// ---------------------------------------------------------------------------
// Coordinate mapping per conv flavor.
// MODE 0: stride-1 conv with reflect padding PAD (stem / head, always valid).
// MODE 1: stride-2 conv, SAME (pad_lo = 0 for k=3,s=2 with even sizes).
// MODE 2: stride-2 transposed conv, SAME (lhs dilation 2, pad_lo = 1).
// ---------------------------------------------------------------------------
template<int MODE>
__device__ __forceinline__ bool map_coord(int o, int kk, int S, int PAD, int& i) {
  if (MODE == 0) {
    int t = o + kk - PAD;
    t = (t < 0) ? -t : t;
    t = (t >= S) ? (2 * S - 2 - t) : t;
    i = t;
    return true;
  } else if (MODE == 1) {
    int t = 2 * o + kk - PAD;
    i = t;
    return (unsigned)t < (unsigned)S;
  } else {
    int t = o + kk - 1;           // dilated-input coordinate
    if (t & 1) return false;      // zero inserted between inputs (also t==-1)
    int h = t >> 1;
    i = h;
    return (unsigned)h < (unsigned)S;
  }
}

// ---------------------------------------------------------------------------
// Implicit-GEMM conv.
// Block = 8 waves, all sharing one N-group (TG), covering 8 consecutive M
// tiles. Per K-chunk the block's NT weight tiles (NT KB) are staged into LDS:
// TDM async DMA double-buffered (wave 0 issues, s_wait_tensorcnt + barrier),
// or a cooperative synchronous copy when the builtin is unavailable.
// A fragment gathered from global (L2-resident bf16), reused across NT WMMAs.
// ---------------------------------------------------------------------------
template<int MODE, bool IN_F32, bool OUT_F32, bool DO_TANH,
         int KSZ, int CI, int NT, int TN>
__global__ void __launch_bounds__(256)
conv_wmma(const void* __restrict__ in_v,
          const __bf16* __restrict__ wpk,
          const float* __restrict__ bias,
          void* __restrict__ out_v,
          int IH, int IW,
          int owShift, int hwShift,
          int CO, int PAD) {
  constexpr int K       = KSZ * KSZ * CI;
  constexpr int NCH     = (K + 31) / 32;
  constexpr int GROUPSN = TN / NT;

  __shared__ __bf16 ls[2 * NT * 512];      // double-buffered B tiles (NT KB each)

  const int tg   = blockIdx.x % GROUPSN;   // constant divisor
  const int bm   = blockIdx.x / GROUPSN;
  const int wIb  = threadIdx.x >> 5;       // wave in block: 0..7
  const int tm   = bm * 8 + wIb;
  const int lane = threadIdx.x & 31;
  const int half = lane >> 4;
  const int l16  = lane & 15;
  const int OW   = 1 << owShift;
  const int hw   = 1 << hwShift;

  const float*  in_f = (const float*)in_v;
  const __bf16* in_b = (const __bf16*)in_v;

  // A-row (output pixel) owned by this lane: shift/mask decode
  const int m   = tm * 16 + l16;
  const int bb  = m >> hwShift;
  const int rem = m & (hw - 1);
  const int oy  = rem >> owShift;
  const int ox  = rem & (OW - 1);

  // --- weight staging helpers -------------------------------------------
  auto stage = [&](int chunk, int buf) {
#ifdef HAVE_TDM
    if (threadIdx.x < 32) {                // wave 0 issues the DMA
      const unsigned long long ga = (unsigned long long)(size_t)
          (wpk + (((size_t)chunk * TN + (size_t)tg * NT) * 32) * 16);
      const unsigned int lds_off =
          (unsigned int)(size_t)(&ls[(size_t)buf * NT * 512]);
      const unsigned int n8 = NT * 128;    // payload in 8-byte elements
      // D# group 0: count=1 | lds_addr | global_addr[56:0] | type=2
      u32x4 g0 = { 1u, lds_off, (unsigned int)ga,
                   (unsigned int)((ga >> 32) & 0x01FFFFFFu) | (2u << 30) };
      // D# group 1: data_size=8B, tensor_dim0=n8, tensor_dim1=1,
      //             tile_dim0=n8, tile_dim1/2 unused, dim0_stride=n8
      i32x8 g1 = { (int)(3u << 16),
                   (int)((n8 & 0xFFFFu) << 16),
                   (int)((n8 >> 16) | (1u << 16)),
                   (int)(n8 << 16),
                   0,
                   (int)n8,
                   0, 0 };
      i32x4 gz4 = { 0, 0, 0, 0 };
      i32x8 gz8 = { 0, 0, 0, 0, 0, 0, 0, 0 };
      __builtin_amdgcn_tensor_load_to_lds(g0, g1, gz4, gz4, gz8, 0);
    }
#else
    // cooperative synchronous copy (16B per thread per step)
    const __bf16* src = wpk + (((size_t)chunk * TN + (size_t)tg * NT) * 32) * 16;
    for (int i = threadIdx.x; i < NT * 64; i += 256)
      ((v8bf*)ls)[(size_t)buf * NT * 64 + i] = ((const v8bf*)src)[i];
#endif
  };
  auto wait_stage = [&]() {
#ifdef HAVE_TDM
    if (threadIdx.x < 32) __builtin_amdgcn_s_wait_tensorcnt(0);
#endif
    __syncthreads();
  };

  v8f acc[NT];
#pragma unroll
  for (int t = 0; t < NT; ++t) acc[t] = (v8f){0.f,0.f,0.f,0.f,0.f,0.f,0.f,0.f};

  stage(0, 0);
  for (int kc = 0; kc < NCH; ++kc) {
    const int cur = kc & 1;
    wait_stage();                          // chunk kc ready; prev reads done
    if (kc + 1 < NCH) stage(kc + 1, cur ^ 1);

    // ---- gather A fragment (shared across NT output-channel tiles) ----
    v16bf afrag;
#pragma unroll
    for (int g = 0; g < 2; ++g) {
      const int k0 = kc * 32 + 8 * half + 16 * g;        // 8 consecutive k
      if ((!IN_F32) && ((CI & 7) == 0) && (k0 + 8 <= K)) {
        const int ci0 = k0 % CI;                          // constant modulus
        const int rk  = k0 / CI;
        const int kx  = rk % KSZ;
        const int ky  = rk / KSZ;
        int iy = 0, ix = 0;
        const bool ok = map_coord<MODE>(oy, ky, IH, PAD, iy) &
                        map_coord<MODE>(ox, kx, IW, PAD, ix);
        if (ok) {
          const __bf16* p = in_b + (((size_t)(bb * IH + iy)) * IW + ix) * CI + ci0;
          v8bf v = *(const v8bf*)p;                       // global_load_b128
#pragma unroll
          for (int e2 = 0; e2 < 8; ++e2) afrag[8 * g + e2] = v[e2];
        } else {
#pragma unroll
          for (int e2 = 0; e2 < 8; ++e2) afrag[8 * g + e2] = (__bf16)0.f;
        }
      } else {
#pragma unroll
        for (int e2 = 0; e2 < 8; ++e2) {
          const int e = 8 * g + e2;
          const int k = kc * 32 + kmap(e, half);
          float v = 0.f;
          if (k < K) {
            const int ci = k % CI;
            const int rk = k / CI;
            const int kx = rk % KSZ;
            const int ky = rk / KSZ;
            int iy = 0, ix = 0;
            const bool ok = map_coord<MODE>(oy, ky, IH, PAD, iy) &
                            map_coord<MODE>(ox, kx, IW, PAD, ix);
            if (ok) {
              const size_t idx = (((size_t)(bb * IH + iy)) * IW + ix) * CI + ci;
              v = IN_F32 ? in_f[idx] : (float)in_b[idx];
            }
          }
          afrag[e] = (__bf16)v;
        }
      }
    }

    // ---- NT B fragments from LDS, NT WMMAs against the same A ----
    const __bf16* lbase = &ls[(size_t)cur * NT * 512 + (size_t)lane * 16];
#pragma unroll
    for (int t = 0; t < NT; ++t) {
      v16bf bfrag = *(const v16bf*)(lbase + (size_t)t * 512);   // ds_load_b128 x2
      acc[t] = __builtin_amdgcn_wmma_f32_16x16x32_bf16(false, afrag, false, bfrag,
                                                       (short)0, acc[t], false, false);
    }
  }

  // ---- epilogue: bias (+tanh) and store ----
#pragma unroll
  for (int t = 0; t < NT; ++t) {
    const int n = (tg * NT + t) * 16 + l16;
    if (n < CO) {
      const float bval = bias[n];
#pragma unroll
      for (int r = 0; r < 8; ++r) {
        const int mm = tm * 16 + r + 8 * half;           // C/D layout, ISA 7.12.2
        float v = acc[t][r] + bval;
        if (DO_TANH) v = tanhf(v);
        if (OUT_F32) ((float*)out_v)[(size_t)mm * CO + n] = v;
        else         ((__bf16*)out_v)[(size_t)mm * CO + n] = (__bf16)v;
      }
    }
  }
}

// ---------------------------------------------------------------------------
// GroupNorm(groups == channels), three phases:
//  1) gn_part: coalesced v8bf grid over octets, register accumulators per
//     fixed channel-octet, ds_add_f32 block reduction, global atomic flush.
//  2) gn_finalize: per-(b,c) mean/var -> sA = rsqrt(var+eps)*gamma,
//     sB = beta - mean*sA.
//  3) gn_apply: vectorized in-place x = relu(x*sA + sB).
// ---------------------------------------------------------------------------
__global__ void zero_f32(float* __restrict__ p, int n) {
  int i = blockIdx.x * blockDim.x + threadIdx.x;
  if (i < n) p[i] = 0.f;
}

__global__ void gn_part(const __bf16* __restrict__ x,
                        float* __restrict__ gsum, float* __restrict__ gsq,
                        int C, int S, int chunk) {
  __shared__ float lsum[256], lsq[256];          // C <= 256
  for (int i = threadIdx.x; i < C; i += blockDim.x) { lsum[i] = 0.f; lsq[i] = 0.f; }
  __syncthreads();
  const int b = blockIdx.x / S;
  const int s = blockIdx.x - b * S;
  const size_t base = ((size_t)b * S + s) * chunk;   // octet index (contiguous)
  float rs[8] = {0.f,0.f,0.f,0.f,0.f,0.f,0.f,0.f};
  float rq[8] = {0.f,0.f,0.f,0.f,0.f,0.f,0.f,0.f};
  const int c0 = (threadIdx.x * 8) & (C - 1);        // fixed per thread
  for (int l = threadIdx.x; l < chunk; l += blockDim.x) {
    v8bf v = *(const v8bf*)(x + (base + l) * 8);
#pragma unroll
    for (int e = 0; e < 8; ++e) { const float f = (float)v[e]; rs[e] += f; rq[e] += f * f; }
  }
#pragma unroll
  for (int e = 0; e < 8; ++e) {
    atomicAdd(&lsum[c0 + e], rs[e]);                 // ds_add_f32
    atomicAdd(&lsq [c0 + e], rq[e]);
  }
  __syncthreads();
  for (int i = threadIdx.x; i < C; i += blockDim.x) {
    atomicAdd(&gsum[b * C + i], lsum[i]);
    atomicAdd(&gsq [b * C + i], lsq[i]);
  }
}

__global__ void gn_finalize(const float* __restrict__ gsum, const float* __restrict__ gsq,
                            const float* __restrict__ gamma, const float* __restrict__ beta,
                            float* __restrict__ sA, float* __restrict__ sB,
                            int total, int cMask, float invHW) {
  int i = blockIdx.x * blockDim.x + threadIdx.x;
  if (i >= total) return;
  const int c = i & cMask;
  const float mean = gsum[i] * invHW;
  float var = gsq[i] * invHW - mean * mean;
  var = fmaxf(var, 0.f);
  const float r = rsqrtf(var + 1e-5f);
  const float a = r * gamma[c];
  sA[i] = a;
  sB[i] = beta[c] - mean * a;
}

__global__ void gn_apply(__bf16* __restrict__ x,
                         const float* __restrict__ sA,
                         const float* __restrict__ sB,
                         int total8, int cMask, int hwcShift, int C) {
  int i8 = blockIdx.x * blockDim.x + threadIdx.x;
  if (i8 >= total8) return;
  const size_t base = (size_t)i8 * 8;
  const int b  = (int)(base >> hwcShift);
  const int c0 = (int)(base & (size_t)cMask);   // multiple of 8, C >= 16 pow2
  v8bf v = *(const v8bf*)(x + base);
  v8f  a = *(const v8f*)(sA + (size_t)b * C + c0);
  v8f  s = *(const v8f*)(sB + (size_t)b * C + c0);
  v8bf o;
#pragma unroll
  for (int e = 0; e < 8; ++e) {
    const float f = (float)v[e] * a[e] + s[e];
    o[e] = (__bf16)fmaxf(f, 0.f);
  }
  *(v8bf*)(x + base) = o;
}

// ---------------------------------------------------------------------------
// Instance average pooling: LDS-atomic partial sums (B*NUM_INST=128 segments),
// one global atomic per block per segment, then scatter means.
// ---------------------------------------------------------------------------
__global__ void pool_accum(const float* __restrict__ head, const int* __restrict__ inst,
                           float* __restrict__ sums, float* __restrict__ cnts,
                           int BHW, int hwShift) {
  __shared__ float ls[4 * NUM_INST * 4];
  for (int i = threadIdx.x; i < 4 * NUM_INST * 4; i += blockDim.x) ls[i] = 0.f;
  __syncthreads();
  const int stride = gridDim.x * blockDim.x;
  for (int i = blockIdx.x * blockDim.x + threadIdx.x; i < BHW; i += stride) {
    const int b   = i >> hwShift;
    const int seg = (b * NUM_INST + inst[i]) * 4;
    atomicAdd(&ls[seg + 0], head[(size_t)i * 3 + 0]);
    atomicAdd(&ls[seg + 1], head[(size_t)i * 3 + 1]);
    atomicAdd(&ls[seg + 2], head[(size_t)i * 3 + 2]);
    atomicAdd(&ls[seg + 3], 1.f);
  }
  __syncthreads();
  for (int s = threadIdx.x; s < 4 * NUM_INST; s += blockDim.x) {
    atomicAdd(&sums[s * 3 + 0], ls[s * 4 + 0]);
    atomicAdd(&sums[s * 3 + 1], ls[s * 4 + 1]);
    atomicAdd(&sums[s * 3 + 2], ls[s * 4 + 2]);
    atomicAdd(&cnts[s],         ls[s * 4 + 3]);
  }
}

__global__ void pool_scatter(const int* __restrict__ inst,
                             const float* __restrict__ sums,
                             const float* __restrict__ cnts,
                             float* __restrict__ out, int BHW, int hwShift) {
  int i = blockIdx.x * blockDim.x + threadIdx.x;
  if (i >= BHW) return;
  const int b   = i >> hwShift;
  const int seg = b * NUM_INST + inst[i];
  const float c = fmaxf(cnts[seg], 1.f);
  out[(size_t)i * 3 + 0] = sums[seg * 3 + 0] / c;
  out[(size_t)i * 3 + 1] = sums[seg * 3 + 1] / c;
  out[(size_t)i * 3 + 2] = sums[seg * 3 + 2] / c;
}

// ---------------------------------------------------------------------------
// Host orchestration.
// Input order (setup_inputs insertion order, params pytree in insertion order):
// 0:x 1:inst 2..5:stem(w,b,g,bt) 6..21:down[i](w,b,g,bt) 22..37:up[i](w,b,g,bt)
// 38:head_w 39:head_b
// ---------------------------------------------------------------------------
extern "C" void kernel_launch(void* const* d_in, const int* in_sizes, int n_in,
                              void* d_out, int out_size, void* d_ws, size_t ws_size,
                              hipStream_t stream) {
  (void)in_sizes; (void)n_in; (void)out_size; (void)ws_size;
  auto cdiv = [](long a, long b) { return (int)((a + b - 1) / b); };

  const float* x       = (const float*)d_in[0];
  const int*   inst    = (const int*)  d_in[1];
  const float* stem_w  = (const float*)d_in[2];
  const float* stem_b  = (const float*)d_in[3];
  const float* stem_g  = (const float*)d_in[4];
  const float* stem_bt = (const float*)d_in[5];
  const float *dw[4], *db[4], *dg[4], *dbt[4];
  for (int i = 0; i < 4; ++i) {
    dw[i]  = (const float*)d_in[6 + 4 * i];
    db[i]  = (const float*)d_in[7 + 4 * i];
    dg[i]  = (const float*)d_in[8 + 4 * i];
    dbt[i] = (const float*)d_in[9 + 4 * i];
  }
  const float *uw[4], *ub[4], *ug[4], *ubt[4];
  for (int i = 0; i < 4; ++i) {
    uw[i]  = (const float*)d_in[22 + 4 * i];
    ub[i]  = (const float*)d_in[23 + 4 * i];
    ug[i]  = (const float*)d_in[24 + 4 * i];
    ubt[i] = (const float*)d_in[25 + 4 * i];
  }
  const float* head_w = (const float*)d_in[38];
  const float* head_b = (const float*)d_in[39];

  char* wsb = (char*)d_ws;
  size_t cur = 0;
  auto walloc = [&](size_t bytes) -> void* {
    cur = (cur + 255) & ~(size_t)255;
    void* p = wsb + cur; cur += bytes; return p;
  };

  __bf16* bufA  = (__bf16*)walloc((size_t)4 * 512 * 512 * 16 * 2);
  __bf16* bufB  = (__bf16*)walloc((size_t)4 * 512 * 512 * 16 * 2);
  float*  headb = (float*) walloc((size_t)4 * 512 * 512 * 3 * 4);
  float*  gsum  = (float*) walloc((size_t)2 * 4 * 256 * 4);  // gsum | gsq
  float*  gsq   = gsum + 4 * 256;
  float*  stA   = (float*) walloc((size_t)4 * 256 * 4);
  float*  stB   = (float*) walloc((size_t)4 * 256 * 4);
  float*  sums  = (float*) walloc((size_t)(4 * NUM_INST * 3 + 4 * NUM_INST) * 4);
  float*  cnts  = sums + 4 * NUM_INST * 3;

  const int dCI[4] = {16, 32, 64, 128};
  const int uCI[4] = {256, 128, 64, 32};

  struct Layer { const float* w; int K, CO, nch, nt; __bf16* pk; };
  Layer L[10];
  L[0] = { stem_w, 7 * 7 * 3, 16, 0, 0, nullptr };
  for (int i = 0; i < 4; ++i) L[1 + i] = { dw[i], 9 * dCI[i], 2 * dCI[i], 0, 0, nullptr };
  for (int i = 0; i < 4; ++i) L[5 + i] = { uw[i], 9 * uCI[i], uCI[i] / 2, 0, 0, nullptr };
  L[9] = { head_w, 7 * 7 * 16, 3, 0, 0, nullptr };

  for (int i = 0; i < 10; ++i) {
    L[i].nch = cdiv(L[i].K, 32);
    L[i].nt  = cdiv(L[i].CO, 16);
    L[i].pk  = (__bf16*)walloc((size_t)L[i].nch * L[i].nt * 512 * 2);
    const int tot = L[i].nch * L[i].nt * 512;
    pack_weights<<<cdiv(tot, 256), 256, 0, stream>>>(L[i].w, L[i].pk,
                                                     L[i].K, L[i].CO, L[i].nch, L[i].nt);
  }

  auto ilog2 = [](int v) { int s = 0; while ((1 << s) < v) ++s; return s; };
  auto launch_gn = [&](__bf16* buf, const float* g, const float* bt, int C, int HW) {
    zero_f32<<<cdiv(2 * 4 * 256, 256), 256, 0, stream>>>(gsum, 2 * 4 * 256);
    const int octPerImg = HW * C / 8;
    int S = octPerImg / 2048; if (S < 1) S = 1;
    const int chunk = octPerImg / S;
    gn_part<<<4 * S, 256, 0, stream>>>(buf, gsum, gsq, C, S, chunk);
    gn_finalize<<<cdiv(4 * C, 256), 256, 0, stream>>>(gsum, gsq, g, bt, stA, stB,
                                                      4 * C, C - 1, 1.f / (float)HW);
    const int total8 = 4 * HW * C / 8;
    gn_apply<<<cdiv(total8, 256), 256, 0, stream>>>(buf, stA, stB, total8,
                                                    C - 1, ilog2(HW * C), C);
  };
  // blocks = (tilesM/8) * (TN/NT); tilesM = 4*HW/16 = HW/4 (HW >= 1024)
  auto nblk = [&](int HW, int groupsN) { return (HW >> 5) * groupsN; };

  // ---- stem: 7x7 s1 reflect pad 3, f32 x -> bf16 bufA (16ch, 512^2) ----
  conv_wmma<0, true, false, false, 7, 3, 1, 1>
      <<<nblk(512 * 512, 1), 256, 0, stream>>>(x, L[0].pk, stem_b, bufA,
                                               512, 512, 9, 18, 16, 3);
  launch_gn(bufA, stem_g, stem_bt, 16, 512 * 512);

  // ---- down chain: 3x3 s2 SAME (bufA->B->A->B->A) ----
  conv_wmma<1, false, false, false, 3, 16, 2, 2>
      <<<nblk(256 * 256, 1), 256, 0, stream>>>(bufA, L[1].pk, db[0], bufB,
                                               512, 512, 8, 16, 32, 0);
  launch_gn(bufB, dg[0], dbt[0], 32, 256 * 256);

  conv_wmma<1, false, false, false, 3, 32, 4, 4>
      <<<nblk(128 * 128, 1), 256, 0, stream>>>(bufB, L[2].pk, db[1], bufA,
                                               256, 256, 7, 14, 64, 0);
  launch_gn(bufA, dg[1], dbt[1], 64, 128 * 128);

  conv_wmma<1, false, false, false, 3, 64, 4, 8>
      <<<nblk(64 * 64, 2), 256, 0, stream>>>(bufA, L[3].pk, db[2], bufB,
                                             128, 128, 6, 12, 128, 0);
  launch_gn(bufB, dg[2], dbt[2], 128, 64 * 64);

  conv_wmma<1, false, false, false, 3, 128, 4, 16>
      <<<nblk(32 * 32, 4), 256, 0, stream>>>(bufB, L[4].pk, db[3], bufA,
                                             64, 64, 5, 10, 256, 0);
  launch_gn(bufA, dg[3], dbt[3], 256, 32 * 32);

  // ---- up chain: 3x3 s2 transposed SAME (bufA->B->A->B->A) ----
  conv_wmma<2, false, false, false, 3, 256, 4, 8>
      <<<nblk(64 * 64, 2), 256, 0, stream>>>(bufA, L[5].pk, ub[0], bufB,
                                             32, 32, 6, 12, 128, 1);
  launch_gn(bufB, ug[0], ubt[0], 128, 64 * 64);

  conv_wmma<2, false, false, false, 3, 128, 4, 4>
      <<<nblk(128 * 128, 1), 256, 0, stream>>>(bufB, L[6].pk, ub[1], bufA,
                                               64, 64, 7, 14, 64, 1);
  launch_gn(bufA, ug[1], ubt[1], 64, 128 * 128);

  conv_wmma<2, false, false, false, 3, 64, 2, 2>
      <<<nblk(256 * 256, 1), 256, 0, stream>>>(bufA, L[7].pk, ub[2], bufB,
                                               128, 128, 8, 16, 32, 1);
  launch_gn(bufB, ug[2], ubt[2], 32, 256 * 256);

  conv_wmma<2, false, false, false, 3, 32, 1, 1>
      <<<nblk(512 * 512, 1), 256, 0, stream>>>(bufB, L[8].pk, ub[3], bufA,
                                               256, 256, 9, 18, 16, 1);
  launch_gn(bufA, ug[3], ubt[3], 16, 512 * 512);

  // ---- head: 7x7 s1 reflect pad 3, bf16 bufA -> f32 headb, tanh ----
  conv_wmma<0, false, true, true, 7, 16, 1, 1>
      <<<nblk(512 * 512, 1), 256, 0, stream>>>(bufA, L[9].pk, head_b, headb,
                                               512, 512, 9, 18, 3, 3);

  // ---- instance average pooling ----
  {
    const int BHW = 4 * 512 * 512;
    zero_f32<<<cdiv(4 * NUM_INST * 4, 256), 256, 0, stream>>>(sums, 4 * NUM_INST * 4);
    pool_accum<<<2048, 256, 0, stream>>>(headb, inst, sums, cnts, BHW, 18);
    pool_scatter<<<cdiv(BHW, 256), 256, 0, stream>>>(inst, sums, cnts,
                                                     (float*)d_out, BHW, 18);
  }
}
